// SelfAttention_33234456936606
// MI455X (gfx1250) — matrix operand order repaired and checked by reading, not compiled, not run on hardware
//
#include <hip/hip_runtime.h>
#include <hip/hip_bf16.h>

// B=4, S=2048, D=1024 causal self-attention for gfx1250 (MI455X).
// All GEMMs via v_wmma_f32_16x16x32_bf16 (fp32 accum). 128x128x32 block tiles,
// 8 waves each computing 32x64 (8 WMMAs/K-step). Global->LDS staging uses
// GLOBAL_LOAD_ASYNC_TO_LDS_B128 (ASYNCcnt) -> no staging VGPRs, no spills;
// double-buffered LDS, one barrier per K-step.

#define BATCH  4
#define SEQ    2048
#define DMODEL 1024

typedef __bf16 bf16_t;
typedef bf16_t v16bf __attribute__((ext_vector_type(16)));
typedef float  v8f   __attribute__((ext_vector_type(8)));
typedef int    v4i_g __attribute__((vector_size(16)));   // matches builtin's V4i

#if defined(__has_builtin)
#if __has_builtin(__builtin_amdgcn_global_load_async_to_lds_b128)
#define HAVE_ASYNC_LDS 1
#endif
#endif
#ifndef HAVE_ASYNC_LDS
#define HAVE_ASYNC_LDS 0
#endif

#define AS_GLOBAL_V4I(p) ((__attribute__((address_space(1))) v4i_g*)(p))
#define AS_LOCAL_V4I(p)  ((__attribute__((address_space(3))) v4i_g*)(p))

__device__ __forceinline__ void wait_async0() {
#if HAVE_ASYNC_LDS
#if __has_builtin(__builtin_amdgcn_s_wait_asynccnt)
    __builtin_amdgcn_s_wait_asynccnt(0);
#else
    asm volatile("s_wait_asynccnt 0x0" ::: "memory");
#endif
#endif
}

union FragBF {
    uint4 q[2];
    v16bf v;
};

__device__ __forceinline__ unsigned short f2bf(float f) {
    // round-to-nearest-even fp32 -> bf16
    unsigned int u = __float_as_uint(f);
    unsigned int r = (u + 0x7FFFu + ((u >> 16) & 1u)) >> 16;
    return (unsigned short)r;
}

// ---------------------------------------------------------------------------
// Elementwise fp32 -> bf16 convert
// ---------------------------------------------------------------------------
__global__ __launch_bounds__(256) void k_f32_to_bf16(const float* __restrict__ src,
                                                     unsigned short* __restrict__ dst,
                                                     int n) {
    int idx = blockIdx.x * 256 + threadIdx.x;
    if (idx < n) dst[idx] = f2bf(src[idx]);
}

// W [in=D][out=D] fp32  ->  Wt [out][in] bf16 (GEMM B operand is [N][K])
__global__ __launch_bounds__(256) void k_transpose_w_bf16(const float* __restrict__ W,
                                                          unsigned short* __restrict__ Wt) {
    int idx = blockIdx.x * 256 + threadIdx.x;   // over D*D, coalesced writes
    int o = idx >> 10;          // / DMODEL
    int i = idx & (DMODEL - 1); // % DMODEL
    Wt[idx] = f2bf(W[(size_t)i * DMODEL + o]);
}

// ---------------------------------------------------------------------------
// Tiled WMMA GEMM:  Out[M][N] = A[M][K] * Bt[N][K]^T  (+ bias[N]); A,Bt bf16.
//   OUT_MODE:    0 = bf16 [M][N], 1 = bf16 V-transpose ([B][D][S]), 2 = fp32 [M][N]
//   CAUSAL_SKIP: skip tiles entirely above the causal diagonal (QK^T)
//   TRUNC_K:     limit K loop to m0+BM (PV: keys <= last query of tile)
// Block: 256 threads (8 waves). Tile BM=128, BN=128, BK=32.
// Wave (wm,wn): rows wm*32..+31, cols wn*64..+63 -> 2x4 WMMA tiles.
// ---------------------------------------------------------------------------
template<int OUT_MODE, int CAUSAL_SKIP, int TRUNC_K>
__global__ __launch_bounds__(256) void k_gemm_wmma(
    const unsigned short* __restrict__ A, int ldA,
    const unsigned short* __restrict__ Bt, int ldBt,
    const float* __restrict__ bias,
    void* __restrict__ Outp, int ldOut,
    int M, int N, int Kdim, float scale)
{
    constexpr int BM = 128, BN = 128, BK = 32;
    const int m0 = blockIdx.y * BM;
    const int n0 = blockIdx.x * BN;
    if (CAUSAL_SKIP && n0 > m0 + BM - 1) return;   // fully-masked tile

    __shared__ __align__(16) unsigned short lA[2][BM * BK];  // [buf][row][k]
    __shared__ __align__(16) unsigned short lB[2][BN * BK];  // [buf][n][k]

    const int tid  = threadIdx.x;
    const int lane = tid & 31;
    const int wid  = tid >> 5;
    const int wm   = wid & 3;     // 0..3 -> 32-row strip
    const int wn   = wid >> 2;    // 0..1 -> 64-col strip
    const int half = lane >> 4;   // 0|1
    const int l16  = lane & 15;

    v8f acc[2][4] = {};

    const int kEnd = TRUNC_K ? min(Kdim, m0 + BM) : Kdim;
    const int nK   = kEnd / BK;

    // per-thread staging coordinates (two 16B chunks each for A and B)
    const int rowS = tid >> 2;            // 0..63  (+64 for second chunk)
    const int segS = tid & 3;             // 16B segment within 64B row

#if HAVE_ASYNC_LDS
    auto fetchTile = [&](int buf, int k0) {
#pragma unroll
        for (int i = 0; i < 2; ++i) {
            int row = rowS + i * 64;
            __builtin_amdgcn_global_load_async_to_lds_b128(
                AS_GLOBAL_V4I(&A[(size_t)(m0 + row) * ldA + k0 + segS * 8]),
                AS_LOCAL_V4I(&lA[buf][row * BK + segS * 8]), 0, 0);
            __builtin_amdgcn_global_load_async_to_lds_b128(
                AS_GLOBAL_V4I(&Bt[(size_t)(n0 + row) * ldBt + k0 + segS * 8]),
                AS_LOCAL_V4I(&lB[buf][row * BK + segS * 8]), 0, 0);
        }
    };
#else
    uint4 aReg[2], bReg[2];
    auto fetchTile = [&](int /*buf*/, int k0) {
#pragma unroll
        for (int i = 0; i < 2; ++i) {
            int row = rowS + i * 64;
            aReg[i] = *(const uint4*)&A[(size_t)(m0 + row) * ldA + k0 + segS * 8];
            bReg[i] = *(const uint4*)&Bt[(size_t)(n0 + row) * ldBt + k0 + segS * 8];
        }
    };
    auto commitTile = [&](int buf) {
#pragma unroll
        for (int i = 0; i < 2; ++i) {
            int row = rowS + i * 64;
            *(uint4*)&lA[buf][row * BK + segS * 8] = aReg[i];
            *(uint4*)&lB[buf][row * BK + segS * 8] = bReg[i];
        }
    };
#endif

    auto compute = [&](int buf) {
        // Fragments per ISA 7.12.2 (wave32, 16-bit):
        // A: lanes 0-15 K={0..7,16..23}; lanes 16-31 K={8..15,24..31}
        // B: lanes 0-15 K={0..7,16..23}; lanes 16-31 K={8..15,24..31}
        FragBF fa[2], fb[4];
#pragma unroll
        for (int mi = 0; mi < 2; ++mi) {
            const unsigned short* ap = &lA[buf][(wm * 32 + mi * 16 + l16) * BK + half * 8];
            fa[mi].q[0] = *(const uint4*)(ap);
            fa[mi].q[1] = *(const uint4*)(ap + 16);
        }
#pragma unroll
        for (int ni = 0; ni < 4; ++ni) {
            const unsigned short* bp = &lB[buf][(wn * 64 + ni * 16 + l16) * BK + half * 8];
            fb[ni].q[0] = *(const uint4*)(bp);
            fb[ni].q[1] = *(const uint4*)(bp + 16);
        }
#pragma unroll
        for (int mi = 0; mi < 2; ++mi)
#pragma unroll
            for (int ni = 0; ni < 4; ++ni)
                acc[mi][ni] = __builtin_amdgcn_wmma_f32_16x16x32_bf16(
                    false, fa[mi].v, false, fb[ni].v, (short)0, acc[mi][ni], false, false);
    };

    // ---- software pipeline: fetch k+1 while computing k ----
#if HAVE_ASYNC_LDS
    fetchTile(0, 0);
    wait_async0();
    __syncthreads();
    for (int kt = 0; kt < nK; ++kt) {
        const int  cur  = kt & 1;
        const bool more = (kt + 1) < nK;
        if (more) fetchTile(cur ^ 1, (kt + 1) * BK);   // async DMA into other buffer
        compute(cur);
        if (more) wait_async0();
        __syncthreads();
    }
#else
    fetchTile(0, 0);
    commitTile(0);
    __syncthreads();
    for (int kt = 0; kt < nK; ++kt) {
        const int  cur  = kt & 1;
        const bool more = (kt + 1) < nK;
        if (more) fetchTile(cur ^ 1, (kt + 1) * BK);
        compute(cur);
        if (more) commitTile(cur ^ 1);
        __syncthreads();
    }
#endif

    // ---- epilogue: C/D layout lanes 0-15 -> M 0..7, lanes 16-31 -> M 8..15 ----
    const int mW = m0 + wm * 32;
    const int nW = n0 + wn * 64;
#pragma unroll
    for (int mi = 0; mi < 2; ++mi) {
#pragma unroll
        for (int ni = 0; ni < 4; ++ni) {
            const v8f a = acc[mi][ni];
            const int n = nW + ni * 16 + l16;
            const float bvv = bias ? bias[n] : 0.0f;
#pragma unroll
            for (int r = 0; r < 8; ++r) {
                float val = a[r] * scale + bvv;
                int m = mW + mi * 16 + half * 8 + r;
                if (OUT_MODE == 2) {
                    ((float*)Outp)[(size_t)m * ldOut + n] = val;
                } else if (OUT_MODE == 0) {
                    ((unsigned short*)Outp)[(size_t)m * ldOut + n] = f2bf(val);
                } else { // V-transpose: m = b*SEQ + s -> Vt[b][n][s]
                    int b = m >> 11;
                    int s = m & (SEQ - 1);
                    ((unsigned short*)Outp)[(size_t)b * DMODEL * SEQ + (size_t)n * SEQ + s] = f2bf(val);
                }
            }
        }
    }
}

// ---------------------------------------------------------------------------
// In-place causal scaled softmax on attn [B][S][S]; one block per row.
// Also emits a bf16 copy of the weights for the PV WMMA GEMM.
// ---------------------------------------------------------------------------
__global__ __launch_bounds__(256) void k_softmax_causal(float* __restrict__ attn,
                                                        unsigned short* __restrict__ attnBf) {
    const int row = blockIdx.x;            // 0 .. B*S-1
    const int q   = row & (SEQ - 1);
    float* p = attn + (size_t)row * SEQ;
    unsigned short* pb = attnBf + (size_t)row * SEQ;
    const float scale = 0.03125f;          // 1/sqrt(1024)
    __shared__ float red[256];
    const int tid = threadIdx.x;

    float m = -3.4e38f;
    for (int n = tid; n <= q; n += 256) m = fmaxf(m, p[n]);
    red[tid] = m;
    __syncthreads();
    for (int s = 128; s > 0; s >>= 1) {
        if (tid < s) red[tid] = fmaxf(red[tid], red[tid + s]);
        __syncthreads();
    }
    const float mx = red[0] * scale;
    __syncthreads();

    float sum = 0.0f;
    for (int n = tid; n <= q; n += 256) sum += __expf(p[n] * scale - mx);
    red[tid] = sum;
    __syncthreads();
    for (int s = 128; s > 0; s >>= 1) {
        if (tid < s) red[tid] += red[tid + s];
        __syncthreads();
    }
    const float inv = 1.0f / red[0];

    for (int n = tid; n < SEQ; n += 256) {
        float w = (n <= q) ? __expf(p[n] * scale - mx) * inv : 0.0f;
        p[n]  = w;
        pb[n] = f2bf(w);
    }
}

// ---------------------------------------------------------------------------
extern "C" void kernel_launch(void* const* d_in, const int* in_sizes, int n_in,
                              void* d_out, int out_size, void* d_ws, size_t ws_size,
                              hipStream_t stream) {
    (void)in_sizes; (void)n_in; (void)out_size; (void)ws_size;
    const float* x  = (const float*)d_in[0];
    // d_in[1] = mask (causal, known statically) -- unused
    const float* Wq = (const float*)d_in[2];
    const float* bq = (const float*)d_in[3];
    const float* Wk = (const float*)d_in[4];
    const float* bk = (const float*)d_in[5];
    const float* Wv = (const float*)d_in[6];
    const float* bv = (const float*)d_in[7];
    const float* Wo = (const float*)d_in[8];
    const float* bo = (const float*)d_in[9];

    float* out  = (float*)d_out;                              // [B,S,D]
    float* attn = out + (size_t)BATCH * SEQ * DMODEL;         // [B,S,S]

    const size_t MD = (size_t)BATCH * SEQ * DMODEL;           // 8 Mi elems
    const size_t DD = (size_t)DMODEL * DMODEL;
    unsigned short* xbf = (unsigned short*)d_ws;  // x bf16; reused as attn_out
    unsigned short* Qbf = xbf + MD;               // Q bf16; later attn_bf (lo half)
    unsigned short* Kbf = Qbf + MD;               // K bf16; later attn_bf (hi half)
    unsigned short* Vt  = Kbf + MD;               // [B][D][S]
    unsigned short* Wqt = Vt  + MD;
    unsigned short* Wkt = Wqt + DD;
    unsigned short* Wvt = Wkt + DD;
    unsigned short* Wot = Wvt + DD;
    unsigned short* attnBf = Qbf;                 // [B][S][S] bf16 = Qbf..Kbf (16 Mi)

    const int MS = BATCH * SEQ;                   // 8192 rows

    // --- precision conversion ---
    k_f32_to_bf16<<<(int)(MD / 256), 256, 0, stream>>>(x, xbf, (int)MD);
    k_transpose_w_bf16<<<(int)(DD / 256), 256, 0, stream>>>(Wq, Wqt);
    k_transpose_w_bf16<<<(int)(DD / 256), 256, 0, stream>>>(Wk, Wkt);
    k_transpose_w_bf16<<<(int)(DD / 256), 256, 0, stream>>>(Wv, Wvt);
    k_transpose_w_bf16<<<(int)(DD / 256), 256, 0, stream>>>(Wo, Wot);

    // --- Q/K/V projections ---
    dim3 gProj(DMODEL / 128, MS / 128);
    k_gemm_wmma<0, 0, 0><<<gProj, 256, 0, stream>>>(
        xbf, DMODEL, Wqt, DMODEL, bq, Qbf, DMODEL, MS, DMODEL, DMODEL, 1.0f);
    k_gemm_wmma<0, 0, 0><<<gProj, 256, 0, stream>>>(
        xbf, DMODEL, Wkt, DMODEL, bk, Kbf, DMODEL, MS, DMODEL, DMODEL, 1.0f);
    k_gemm_wmma<1, 0, 0><<<gProj, 256, 0, stream>>>(
        xbf, DMODEL, Wvt, DMODEL, bv, Vt, 0, MS, DMODEL, DMODEL, 1.0f);

    // --- scores = Q K^T (causal tile skip), raw fp32 into attn region ---
    dim3 gScore(SEQ / 128, SEQ / 128);
    for (int b = 0; b < BATCH; ++b) {
        k_gemm_wmma<2, 1, 0><<<gScore, 256, 0, stream>>>(
            Qbf + (size_t)b * SEQ * DMODEL, DMODEL,
            Kbf + (size_t)b * SEQ * DMODEL, DMODEL,
            nullptr,
            attn + (size_t)b * SEQ * SEQ, SEQ,
            SEQ, SEQ, DMODEL, 1.0f);
    }

    // --- masked scaled softmax in place (+ bf16 copy into dead Q/K space) ---
    k_softmax_causal<<<MS, 256, 0, stream>>>(attn, attnBf);

    // --- attn_out = P V  (K truncated to causal extent) ---
    dim3 gPV(DMODEL / 128, SEQ / 128);
    for (int b = 0; b < BATCH; ++b) {
        k_gemm_wmma<0, 0, 1><<<gPV, 256, 0, stream>>>(
            attnBf + (size_t)b * SEQ * SEQ, SEQ,
            Vt + (size_t)b * DMODEL * SEQ, SEQ,
            nullptr,
            xbf + (size_t)b * SEQ * DMODEL, DMODEL,
            SEQ, DMODEL, SEQ, 1.0f);
    }

    // --- output projection (fp32 out) ---
    k_gemm_wmma<2, 0, 0><<<gProj, 256, 0, stream>>>(
        xbf, DMODEL, Wot, DMODEL, bo, out, DMODEL, MS, DMODEL, DMODEL, 1.0f);
}
